// Memory_6554120093949
// MI455X (gfx1250) — matrix operand order repaired and checked by reading, not compiled, not run on hardware
//
#include <hip/hip_runtime.h>

// ---------------------------------------------------------------------------
// Problem constants (from reference): B=64 queries, Dfeat=256, C=345 (pad 352),
// N=200000 memory rows, TOP=5, MOMENTUM=0.1.
// ---------------------------------------------------------------------------
#define BQ      64
#define DF      256
#define CC      345
#define CP      352      // 345 padded to multiple of 32 for bf16 WMMA K
#define NROWS   200000
#define CH_IT   13       // 13 * 32 = 416 rows per workgroup
#define NWG     481      // 481 * 416 = 200096 >= N
#define NCQ     (NWG * 8) // candidates per query from streaming pass
#define NEG_INF (-3.0e38f)
#define EPS_N   1e-12f

typedef __bf16 v16bf __attribute__((ext_vector_type(16)));
typedef float  v8f   __attribute__((ext_vector_type(8)));
typedef unsigned int u32x4 __attribute__((ext_vector_type(4)));
typedef int          i32x4 __attribute__((ext_vector_type(4)));
typedef int          i32x8 __attribute__((ext_vector_type(8)));

#if defined(__HIP_DEVICE_COMPILE__) && defined(__has_builtin)
#  if __has_builtin(__builtin_amdgcn_tensor_load_to_lds) && \
      __has_builtin(__builtin_amdgcn_s_wait_tensorcnt)
#    define USE_TDM 1
#    warning "ATHENA-PROBE: TDM builtin available -- tensor_load_to_lds path ENABLED"
#  else
#    warning "ATHENA-PROBE: TDM builtin NOT available -- global-load fallback path"
#  endif
#endif
#ifndef USE_TDM
#  define USE_TDM 0
#endif

#if USE_TDM
// ---------------------------------------------------------------------------
// Issue one TDM 2D tile load: 32 x D fp32 elements starting at gptr into
// LDS offset lds_off (packed row-major, stride D). Rows past the tensor edge
// (tensor_dim1 = rows_remaining) are zero-filled by TDM OOB semantics.
// D# bit layout per CDNA5 ISA §8.3/§8.4 (group0 128b, group1 256b).
// ---------------------------------------------------------------------------
__device__ __forceinline__ void tdm_load_tile(const float* gptr, unsigned lds_off,
                                              int rows_remaining, int D, int tile_rows)
{
  const unsigned long long ga = (unsigned long long)(__UINTPTR_TYPE__)gptr;
  const unsigned td1 = (rows_remaining > 0) ? (unsigned)rows_remaining : 0u;
  const unsigned uD = (unsigned)D;

  u32x4 g0;
  g0[0] = 1u;                                            // count=1 (valid), user mode
  g0[1] = lds_off;                                       // lds_addr [63:32]
  g0[2] = (unsigned)(ga & 0xffffffffull);                // global_addr [95:64]
  g0[3] = (unsigned)((ga >> 32) & 0x01ffffffull)         // global_addr [120:96]
          | (2u << 30);                                  // type=2 ("image")

  i32x8 g1;
  g1[0] = (int)(2u << 16);                               // data_size=2 (4B), no flags
  g1[1] = (int)((uD & 0xffffu) << 16);                   // tensor_dim0[15:0] @ [63:48]
  g1[2] = (int)(((uD >> 16) & 0xffffu)                   // tensor_dim0[31:16]
          | ((td1 & 0xffffu) << 16));                    // tensor_dim1[15:0]
  g1[3] = (int)(((td1 >> 16) & 0xffffu)                  // tensor_dim1[31:16]
          | ((uD & 0xffffu) << 16));                     // tile_dim0 = D
  g1[4] = (int)((unsigned)tile_rows & 0xffffu);          // tile_dim1 = rows, tile_dim2=0
  g1[5] = (int)uD;                                       // tensor_dim0_stride[31:0]
  g1[6] = 0;                                             // stride hi / dim1_stride lo
  g1[7] = 0;

  i32x4 gz = {0, 0, 0, 0};
#  if __clang_major__ >= 23
  i32x8 gz8 = {0, 0, 0, 0, 0, 0, 0, 0};
  __builtin_amdgcn_tensor_load_to_lds(g0, g1, gz, gz, gz8, 0);
#  else
  __builtin_amdgcn_tensor_load_to_lds(g0, g1, gz, gz, 0);
#  endif
}
#endif // USE_TDM

// ---------------------------------------------------------------------------
// Kernel 1: softmax, normalized (padded) queries, momentum-updated rows +
// inverse norms, override flag map. Single block, threads 0..63 own a row.
// ---------------------------------------------------------------------------
__global__ void __launch_bounds__(256)
prep_kernel(const float* __restrict__ tf, const float* __restrict__ fc1,
            const float* __restrict__ fc2, const int* __restrict__ idxv,
            const float* __restrict__ featmem, const float* __restrict__ fc1mem,
            const float* __restrict__ fc2mem,
            float* __restrict__ qf, float* __restrict__ q1, float* __restrict__ q2,
            float* __restrict__ fc1s, float* __restrict__ fc2s,
            float* __restrict__ featU, float* __restrict__ featUinv,
            float* __restrict__ fc1U, float* __restrict__ fc1Uinv,
            float* __restrict__ fc2U, float* __restrict__ fc2Uinv,
            int* __restrict__ flag)
{
  const int tid = threadIdx.x;
  for (int i = tid; i < NROWS; i += 256) flag[i] = 0;
  __syncthreads();
  if (tid < BQ) {
    const int b = tid;
    {
      const float* x = fc1 + b * CC;
      float mx = x[0];
      for (int k = 1; k < CC; ++k) mx = fmaxf(mx, x[k]);
      float sum = 0.f;
      for (int k = 0; k < CC; ++k) { float e = __expf(x[k] - mx); fc1s[b*CC+k] = e; sum += e; }
      float inv = 1.0f / sum;
      for (int k = 0; k < CC; ++k) fc1s[b*CC+k] *= inv;
    }
    {
      const float* x = fc2 + b * CC;
      float mx = x[0];
      for (int k = 1; k < CC; ++k) mx = fmaxf(mx, x[k]);
      float sum = 0.f;
      for (int k = 0; k < CC; ++k) { float e = __expf(x[k] - mx); fc2s[b*CC+k] = e; sum += e; }
      float inv = 1.0f / sum;
      for (int k = 0; k < CC; ++k) fc2s[b*CC+k] *= inv;
    }
    {
      const float* x = tf + b * DF;
      float s = 0.f;
      for (int k = 0; k < DF; ++k) s += x[k] * x[k];
      float inv = 1.0f / fmaxf(sqrtf(s), EPS_N);
      for (int k = 0; k < DF; ++k) qf[b*DF+k] = x[k] * inv;
    }
    {
      const float* x = fc1s + b * CC;
      float s = 0.f;
      for (int k = 0; k < CC; ++k) s += x[k] * x[k];
      float inv = 1.0f / fmaxf(sqrtf(s), EPS_N);
      for (int k = 0; k < CP; ++k) q1[b*CP+k] = (k < CC) ? x[k] * inv : 0.f;
    }
    {
      const float* x = fc2s + b * CC;
      float s = 0.f;
      for (int k = 0; k < CC; ++k) s += x[k] * x[k];
      float inv = 1.0f / fmaxf(sqrtf(s), EPS_N);
      for (int k = 0; k < CP; ++k) q2[b*CP+k] = (k < CC) ? x[k] * inv : 0.f;
    }
    const int n = idxv[b];
    {
      float s = 0.f;
      for (int k = 0; k < DF; ++k) {
        float v = 0.9f * featmem[(size_t)n*DF + k] + 0.1f * tf[b*DF + k];
        featU[b*DF + k] = v; s += v * v;
      }
      featUinv[b] = 1.0f / fmaxf(sqrtf(s), EPS_N);
    }
    {
      float s = 0.f;
      for (int k = 0; k < CP; ++k) {
        float v = (k < CC) ? (0.9f * fc1mem[(size_t)n*CC + k] + 0.1f * fc1s[b*CC + k]) : 0.f;
        fc1U[b*CP + k] = v; s += v * v;
      }
      fc1Uinv[b] = 1.0f / fmaxf(sqrtf(s), EPS_N);
    }
    {
      float s = 0.f;
      for (int k = 0; k < CP; ++k) {
        float v = (k < CC) ? (0.9f * fc2mem[(size_t)n*CC + k] + 0.1f * fc2s[b*CC + k]) : 0.f;
        fc2U[b*CP + k] = v; s += v * v;
      }
      fc2Uinv[b] = 1.0f / fmaxf(sqrtf(s), EPS_N);
    }
    flag[n] = b + 1;   // inverse map: bank row n is overridden by update row b
  }
}

// ---------------------------------------------------------------------------
// Kernel 2: streaming bf16-WMMA similarity + per-WG top-8 per query.
// Templated on <D, DPAD>; K-loop fully unrolled into TWO independent
// accumulator chains so ds_load_b128s overlap v_wmma via partial dscnt waits
// and the XDL pipe sees two dependency chains instead of one.
// TDM double-buffers fp32 tiles HBM->LDS (wave 0 issues, TENSORcnt tracks).
// ---------------------------------------------------------------------------
template <int D, int DPAD>
__global__ void __launch_bounds__(256)
sim_topk_kernel(const float* __restrict__ mem, const float* __restrict__ qn,
                float* __restrict__ cand_val, int* __restrict__ cand_idx,
                int N, int iters, int ncq)
{
  extern __shared__ char smem[];
  __bf16* sQ       = (__bf16*)smem;                 // [64][DPAD]
  __bf16* sB       = sQ + 64 * DPAD;                // [32][DPAD]
  float*  sNormInv = (float*)(sB + 32 * DPAD);      // [32]
  float*  sSim     = sNormInv + 32;                 // [64][32]
  float*  stage    = sSim + 64 * 32;                // [2][32][D] fp32 staging (TDM)

  const int tid  = threadIdx.x;
  const int wave = tid >> 5, lane = tid & 31;
  const int l = lane & 15, h = lane >> 4;
  const int mq = (wave & 3) << 4;   // query tile base (0/16/32/48)
  const int nb = (wave >> 2) << 4;  // row tile base within 32 (0/16)

  // stage normalized queries to bf16 LDS
  {
    const int r = tid >> 2, j = tid & 3;
    for (int k = j; k < DPAD; k += 4)
      sQ[r * DPAD + k] = (__bf16)qn[r * DPAD + k];
  }

  // running top-8 (thread t < 64 owns query t)
  float tv[8]; int tix[8];
#pragma unroll
  for (int i = 0; i < 8; ++i) { tv[i] = NEG_INF; tix[i] = -1; }

  const long long base0 = (long long)blockIdx.x * iters * 32;

#if USE_TDM
  const unsigned stage_lds = (unsigned)(__UINTPTR_TYPE__)(void*)stage;
  const unsigned stage_bytes = (unsigned)(32 * D * 4);
  if (wave == 0) {  // prologue: tile 0 -> buffer 0
    const long long t0 = base0;
    const float* g = mem + (size_t)(t0 < (long long)N ? t0 : 0) * D;
    tdm_load_tile(g, stage_lds, (int)((long long)N - t0), D, 32);
  }
#endif

  for (int it = 0; it < iters; ++it) {
    const long long tbase = base0 + (long long)it * 32;
    __syncthreads();                       // previous scan done / buffers free

#if USE_TDM
    if (wave == 0) {
      if (it + 1 < iters) {                // issue next tile into other buffer
        const long long tn = tbase + 32;
        const float* g = mem + (size_t)(tn < (long long)N ? tn : 0) * D;
        tdm_load_tile(g, stage_lds + ((unsigned)(it + 1) & 1u) * stage_bytes,
                      (int)((long long)N - tn), D, 32);
        __builtin_amdgcn_s_wait_tensorcnt(1);  // current tile complete
      } else {
        __builtin_amdgcn_s_wait_tensorcnt(0);
      }
    }
    __syncthreads();                       // staged tile visible to all waves
    // convert fp32 staging -> bf16 sB, fp32 row norms
    {
      const int r = tid >> 3, j = tid & 7;
      const float* srow = stage + (size_t)(it & 1) * 32 * D + (size_t)r * D;
      float sq = 0.f;
#pragma unroll 4
      for (int k = j; k < DPAD; k += 8) {
        float v = (k < D) ? srow[k] : 0.f;   // OOB rows already zero-filled by TDM
        sB[r * DPAD + k] = (__bf16)v;
        sq += v * v;
      }
      sq += __shfl_xor(sq, 1, 32);
      sq += __shfl_xor(sq, 2, 32);
      sq += __shfl_xor(sq, 4, 32);
      if (j == 0) sNormInv[r] = 1.0f / fmaxf(sqrtf(sq), EPS_N);
    }
#else
    // fallback: direct global loads + convert
    {
      const int r = tid >> 3, j = tid & 7;
      const long long n = tbase + r;
      const float* row = mem + (size_t)(n < (long long)N ? n : 0) * D;
      float sq = 0.f;
#pragma unroll 4
      for (int k = j; k < DPAD; k += 8) {
        float v = 0.f;
        if (n < (long long)N && k < D) v = row[k];
        sB[r * DPAD + k] = (__bf16)v;
        sq += v * v;
      }
      sq += __shfl_xor(sq, 1, 32);
      sq += __shfl_xor(sq, 2, 32);
      sq += __shfl_xor(sq, 4, 32);
      if (j == 0) sNormInv[r] = 1.0f / fmaxf(sqrtf(sq), EPS_N);
      const long long np = tbase + 32 + r;
      if (np < (long long)N)
        __builtin_prefetch(mem + (size_t)np * D + j * (D >> 3), 0, 0);
    }
#endif
    __syncthreads();

    // 16x16 bf16 WMMA tile per wave; fully unrolled, two accumulator chains
    v8f acc0 = {}, acc1 = {};
    const __bf16* qrow = sQ + (mq + l) * DPAD;
    const __bf16* brow = sB + (nb + l) * DPAD;
    auto load_frag = [&](v16bf& av, v16bf& bv, int kbase) {
#pragma unroll
      for (int j = 0; j < 8; ++j) {
        const int ka = ((j < 4) ? (2 * j) : (16 + 2 * (j - 4))) + (h << 3); // A layout
        const int kb = ((j < 4) ? (2 * j) : (16 + 2 * (j - 4))) + (h << 3); // B layout
        av[2*j]   = qrow[kbase + ka];
        av[2*j+1] = qrow[kbase + ka + 1];
        bv[2*j]   = brow[kbase + kb];
        bv[2*j+1] = brow[kbase + kb + 1];
      }
    };
#pragma unroll
    for (int ks = 0; ks < DPAD; ks += 64) {
      v16bf a0, b0, a1, b1;
      const bool second = (ks + 32 < DPAD);
      load_frag(a0, b0, ks);                       // chain-0 fragment
      if (second) load_frag(a1, b1, ks + 32);      // chain-1 fragment (in flight)
      acc0 = __builtin_amdgcn_wmma_f32_16x16x32_bf16(false, a0, false, b0,
                                                     (short)0, acc0, false, false);
      if (second)
        acc1 = __builtin_amdgcn_wmma_f32_16x16x32_bf16(false, a1, false, b1,
                                                       (short)0, acc1, false, false);
    }

    // write normalized sims: D-layout row m = mq + r + 8h, col n = nb + l
    {
      const float ninv = sNormInv[nb + l];
      const long long n = tbase + nb + l;
      const bool ok = (n < (long long)N);
#pragma unroll
      for (int r = 0; r < 8; ++r) {
        const int m = mq + r + (h << 3);
        sSim[m * 32 + nb + l] = ok ? (acc0[r] + acc1[r]) * ninv : NEG_INF;
      }
    }
    __syncthreads();
    // per-query running top-8 update
    if (tid < BQ) {
#pragma unroll 1
      for (int c = 0; c < 32; ++c) {
        const float v = sSim[tid * 32 + c];
        if (v > tv[7]) {
          const int n = (int)(tbase + c);
          int p = 7;
          while (p > 0 && tv[p-1] < v) { tv[p] = tv[p-1]; tix[p] = tix[p-1]; --p; }
          tv[p] = v; tix[p] = n;
        }
      }
    }
  }
  if (tid < BQ) {
    float* cv = cand_val + (size_t)tid * ncq + blockIdx.x * 8;
    int*   ci = cand_idx + (size_t)tid * ncq + blockIdx.x * 8;
#pragma unroll
    for (int i = 0; i < 8; ++i) { cv[i] = tv[i]; ci[i] = tix[i]; }
  }
}

// ---------------------------------------------------------------------------
// Kernel 3: per-query merge. Exact fp32 sims vs the 64 updated rows, filter
// stale (overridden) candidates, extract global top-Kout by block argmax.
// ---------------------------------------------------------------------------
__global__ void __launch_bounds__(256)
merge_topk_kernel(const float* __restrict__ cand_val, const int* __restrict__ cand_idx,
                  int ncq, const float* __restrict__ qn, int Dpad,
                  const float* __restrict__ upd, const float* __restrict__ updInv,
                  const int* __restrict__ idxv, const int* __restrict__ flag,
                  int N, int Kout, int* __restrict__ topk_out)
{
  __shared__ float sOval[BQ];
  __shared__ float rv[256];
  __shared__ int   ri[256];
  __shared__ int   sChoice;
  const int tid = threadIdx.x;
  const int m = blockIdx.x;

  for (int b = 0; b < BQ; ++b) {
    float p = 0.f;
    for (int k = tid; k < Dpad; k += 256)
      p += qn[m * Dpad + k] * upd[b * Dpad + k];
    rv[tid] = p; __syncthreads();
    for (int off = 128; off > 0; off >>= 1) {
      if (tid < off) rv[tid] += rv[tid + off];
      __syncthreads();
    }
    if (tid == 0) sOval[b] = rv[0] * updInv[b];
    __syncthreads();
  }

  const int S = ncq + BQ;
  int chosen[8];
  for (int j = 0; j < Kout; ++j) {
    float bv = NEG_INF; int bs = -1;
    for (int s = tid; s < S; s += 256) {
      bool skip = false;
      for (int q = 0; q < j; ++q) if (chosen[q] == s) { skip = true; break; }
      if (skip) continue;
      float v; int n;
      if (s < ncq) {
        n = cand_idx[(size_t)m * ncq + s];
        if (n < 0 || n >= N || flag[n] != 0) continue;   // stale/invalid
        v = cand_val[(size_t)m * ncq + s];
      } else {
        v = sOval[s - ncq];
        n = idxv[s - ncq];
      }
      if (v > bv || (v == bv && (unsigned)s < (unsigned)bs)) { bv = v; bs = s; }
    }
    rv[tid] = bv; ri[tid] = bs; __syncthreads();
    for (int off = 128; off > 0; off >>= 1) {
      if (tid < off) {
        if (rv[tid+off] > rv[tid] ||
            (rv[tid+off] == rv[tid] && (unsigned)ri[tid+off] < (unsigned)ri[tid])) {
          rv[tid] = rv[tid+off]; ri[tid] = ri[tid+off];
        }
      }
      __syncthreads();
    }
    if (tid == 0) {
      const int s = ri[0];
      sChoice = s;
      int n = 0;
      if (s >= 0) n = (s < ncq) ? cand_idx[(size_t)m * ncq + s] : idxv[s - ncq];
      topk_out[m * Kout + j] = n;
    }
    __syncthreads();
    chosen[j] = sChoice;
    __syncthreads();
  }
}

// ---------------------------------------------------------------------------
// Loss kernels (effective rows go through the override map).
// ---------------------------------------------------------------------------
__global__ void __launch_bounds__(256)
loss_softmax_kernel(const int* __restrict__ topkF,
                    const float* __restrict__ fc1mem, const float* __restrict__ fc2mem,
                    const float* __restrict__ fc1U, const float* __restrict__ fc2U,
                    const int* __restrict__ flag, float* __restrict__ partial)
{
  __shared__ float red[256];
  const int tid = threadIdx.x;
  const int rrow = blockIdx.x;                 // 0..319 (64 queries x 5 neighbors)
  const int m = rrow / 5, j = rrow % 5 + 1;    // drop first
  const int n = topkF[m * 6 + j];
  const int f = (n >= 0 && n < NROWS) ? flag[n] : 0;
  const float* a = f ? (fc1U + (size_t)(f - 1) * CP) : (fc1mem + (size_t)n * CC);
  const float* b = f ? (fc2U + (size_t)(f - 1) * CP) : (fc2mem + (size_t)n * CC);
  float s = 0.f;
  for (int k = tid; k < CC; k += 256) s += fabsf(a[k] - b[k]);
  red[tid] = s; __syncthreads();
  for (int off = 128; off > 0; off >>= 1) {
    if (tid < off) red[tid] += red[tid + off];
    __syncthreads();
  }
  if (tid == 0) partial[rrow] = red[0];
}

__global__ void __launch_bounds__(256)
loss_feature_kernel(const int* __restrict__ topk1, const int* __restrict__ topk2,
                    const float* __restrict__ featmem, const float* __restrict__ featU,
                    const int* __restrict__ flag, float* __restrict__ partial)
{
  __shared__ float rd[256], r1[256], r2[256];
  const int tid = threadIdx.x;
  const int rrow = blockIdx.x;                 // 0..255 (64 queries x 4 neighbors)
  const int m = rrow >> 2, j = (rrow & 3) + 1; // drop first
  const int n1 = topk1[m * 5 + j];
  const int n2 = topk2[m * 5 + j];
  const int f1 = (n1 >= 0 && n1 < NROWS) ? flag[n1] : 0;
  const int f2 = (n2 >= 0 && n2 < NROWS) ? flag[n2] : 0;
  const float* a = f1 ? (featU + (size_t)(f1 - 1) * DF) : (featmem + (size_t)n1 * DF);
  const float* b = f2 ? (featU + (size_t)(f2 - 1) * DF) : (featmem + (size_t)n2 * DF);
  float d = 0.f, s1 = 0.f, s2 = 0.f;
  for (int k = tid; k < DF; k += 256) {
    const float x = a[k], y = b[k];
    d += x * y; s1 += x * x; s2 += y * y;
  }
  rd[tid] = d; r1[tid] = s1; r2[tid] = s2; __syncthreads();
  for (int off = 128; off > 0; off >>= 1) {
    if (tid < off) { rd[tid] += rd[tid+off]; r1[tid] += r1[tid+off]; r2[tid] += r2[tid+off]; }
    __syncthreads();
  }
  if (tid == 0) {
    const float i1 = 1.0f / fmaxf(sqrtf(r1[0]), EPS_N);
    const float i2 = 1.0f / fmaxf(sqrtf(r2[0]), EPS_N);
    partial[rrow] = 0.5f * (1.0f - rd[0] * i1 * i2);
  }
}

__global__ void __launch_bounds__(256)
finalize_kernel(const float* __restrict__ pS, const float* __restrict__ pF,
                float* __restrict__ out)
{
  __shared__ float red[256];
  const int tid = threadIdx.x;
  float s = 0.f;
  for (int i = tid; i < 320; i += 256) s += pS[i];
  red[tid] = s; __syncthreads();
  for (int off = 128; off > 0; off >>= 1) {
    if (tid < off) red[tid] += red[tid + off];
    __syncthreads();
  }
  if (tid == 0) out[0] = red[0] / (320.0f * 345.0f);
  __syncthreads();
  red[tid] = pF[tid]; __syncthreads();
  for (int off = 128; off > 0; off >>= 1) {
    if (tid < off) red[tid] += red[tid + off];
    __syncthreads();
  }
  if (tid == 0) out[1] = red[0] / 256.0f;
}

// ---------------------------------------------------------------------------
extern "C" void kernel_launch(void* const* d_in, const int* in_sizes, int n_in,
                              void* d_out, int out_size, void* d_ws, size_t ws_size,
                              hipStream_t stream) {
  const float* tf      = (const float*)d_in[0];
  const float* fc1     = (const float*)d_in[1];
  const float* fc2     = (const float*)d_in[2];
  const int*   idxv    = (const int*)  d_in[3];
  const float* featmem = (const float*)d_in[4];
  const float* fc1mem  = (const float*)d_in[5];
  const float* fc2mem  = (const float*)d_in[6];
  float* out = (float*)d_out;

  char* w = (char*)d_ws;
  auto alloc = [&](size_t bytes) -> void* {
    void* p = (void*)w;
    w += (bytes + 255) & ~(size_t)255;
    return p;
  };
  float* qf       = (float*)alloc((size_t)BQ * DF * 4);
  float* q1       = (float*)alloc((size_t)BQ * CP * 4);
  float* q2       = (float*)alloc((size_t)BQ * CP * 4);
  float* fc1s     = (float*)alloc((size_t)BQ * CC * 4);
  float* fc2s     = (float*)alloc((size_t)BQ * CC * 4);
  float* featU    = (float*)alloc((size_t)BQ * DF * 4);
  float* featUinv = (float*)alloc((size_t)BQ * 4);
  float* fc1U     = (float*)alloc((size_t)BQ * CP * 4);
  float* fc1Uinv  = (float*)alloc((size_t)BQ * 4);
  float* fc2U     = (float*)alloc((size_t)BQ * CP * 4);
  float* fc2Uinv  = (float*)alloc((size_t)BQ * 4);
  int*   flag     = (int*)  alloc((size_t)NROWS * 4);
  float* cand_val = (float*)alloc((size_t)BQ * NCQ * 4);
  int*   cand_idx = (int*)  alloc((size_t)BQ * NCQ * 4);
  int*   topkF    = (int*)  alloc((size_t)BQ * 6 * 4);
  int*   topk1    = (int*)  alloc((size_t)BQ * 5 * 4);
  int*   topk2    = (int*)  alloc((size_t)BQ * 5 * 4);
  float* pS       = (float*)alloc(320 * 4);
  float* pF       = (float*)alloc(256 * 4);

  // LDS: sQ + sB (bf16) + norms + sims + double-buffered fp32 TDM staging
  const size_t smemF = (size_t)(64 + 32) * DF * 2 + 32 * 4 + 64 * 32 * 4
                     + (size_t)2 * 32 * DF * 4;   // 57472 + 65536 B
  const size_t smemC = (size_t)(64 + 32) * CP * 2 + 32 * 4 + 64 * 32 * 4
                     + (size_t)2 * 32 * CC * 4;   // 75904 + 88320 B

  prep_kernel<<<1, 256, 0, stream>>>(tf, fc1, fc2, idxv, featmem, fc1mem, fc2mem,
                                     qf, q1, q2, fc1s, fc2s,
                                     featU, featUinv, fc1U, fc1Uinv, fc2U, fc2Uinv, flag);

  // pass 1: feature bank, top-6
  sim_topk_kernel<DF, DF><<<NWG, 256, smemF, stream>>>(featmem, qf, cand_val, cand_idx,
                                                       NROWS, CH_IT, NCQ);
  merge_topk_kernel<<<BQ, 256, 0, stream>>>(cand_val, cand_idx, NCQ, qf, DF,
                                            featU, featUinv, idxv, flag, NROWS, 6, topkF);
  // pass 2: fc1 bank, top-5
  sim_topk_kernel<CC, CP><<<NWG, 256, smemC, stream>>>(fc1mem, q1, cand_val, cand_idx,
                                                       NROWS, CH_IT, NCQ);
  merge_topk_kernel<<<BQ, 256, 0, stream>>>(cand_val, cand_idx, NCQ, q1, CP,
                                            fc1U, fc1Uinv, idxv, flag, NROWS, 5, topk1);
  // pass 3: fc2 bank, top-5
  sim_topk_kernel<CC, CP><<<NWG, 256, smemC, stream>>>(fc2mem, q2, cand_val, cand_idx,
                                                       NROWS, CH_IT, NCQ);
  merge_topk_kernel<<<BQ, 256, 0, stream>>>(cand_val, cand_idx, NCQ, q2, CP,
                                            fc2U, fc2Uinv, idxv, flag, NROWS, 5, topk2);

  loss_softmax_kernel<<<320, 256, 0, stream>>>(topkF, fc1mem, fc2mem, fc1U, fc2U, flag, pS);
  loss_feature_kernel<<<256, 256, 0, stream>>>(topk1, topk2, featmem, featU, flag, pF);
  finalize_kernel<<<1, 256, 0, stream>>>(pS, pF, out);
  (void)in_sizes; (void)n_in; (void)out_size; (void)ws_size;
}
